// RNN_Model_39960375722364
// MI455X (gfx1250) — matrix operand order repaired and checked by reading, not compiled, not run on hardware
//
#include <hip/hip_runtime.h>
#include <hip/hip_bf16.h>
#include <math.h>

// ---------------------------------------------------------------------------
// Problem constants (match reference)
// ---------------------------------------------------------------------------
#define BATCH 4096
#define TT    300
#define FF    39
#define H1C   35
#define H2C   30
#define IN1   (FF + H1C)    // 74
#define IN2   (H1C + H2C)   // 65
#define N1    (4 * H1C)     // 140
#define N2    (4 * H2C)     // 120
#define K1P   96            // IN1 padded to 3 x 32
#define K2P   96            // IN2 padded to 3 x 32
#define N1T   9             // ceil(140/16)
#define NTAPS 7
#define FC1C  240
#define FC2C  80
#define OUTC  2
#define FCIN  (NTAPS * 2 * H2C)  // 420
#define FCINP 448                // padded to 14 x 32
#define XPF   3                  // ceil(16*39 / 256) prefetch regs per thread

typedef __attribute__((ext_vector_type(16))) _Float16 v16h;
typedef __attribute__((ext_vector_type(8)))  float    v8f;

// ---------------------------------------------------------------------------
// Fast transcendentals: v_exp_f32 + v_rcp_f32, no IEEE div sequences.
// sig(v)  = 1/(1+e^-v)          -> saturates correctly (rcp(inf)=0)
// tanh(v) = 1 - 2/(1+e^{2v})    -> +-1 at large |v|
// ---------------------------------------------------------------------------
__device__ __forceinline__ float fast_rcp(float x) {
    return __builtin_amdgcn_rcpf(x);
}
__device__ __forceinline__ float sigf(float v) {
    return fast_rcp(1.0f + __expf(-v));
}
__device__ __forceinline__ float tanhfast(float v) {
    return 1.0f - 2.0f * fast_rcp(1.0f + __expf(2.0f * v));
}

__device__ __forceinline__ int tap_index(int t) {
    switch (t) {
        case 0:   return 0;
        case 49:  return 1;
        case 99:  return 2;
        case 149: return 3;
        case 199: return 4;
        case 249: return 5;
        case 299: return 6;
    }
    return -1;
}

// ---------------------------------------------------------------------------
// Fragment helpers (wave32); A and B fragments share one contiguous k order
// ---------------------------------------------------------------------------

// A fragment: 16x32 f16 tile, source row-major [16][strideH] in LDS.
// lane L holds row m = L&15; half h -> k = kc*32 + (L>>4)*16 + h
__device__ __forceinline__ v16h load_a_frag(const _Float16* As, int lane,
                                            int kc, int strideH) {
    const int m = lane & 15;
    const int kbase = kc * 32 + ((lane >> 4) * 16);
    v16h a;
#pragma unroll
    for (int h = 0; h < 16; ++h) {
        int k = kbase + h;
        a[h] = As[m * strideH + k];
    }
    return a;
}

// B fragment: 32x16 f16 tile from row-major f32 weights W[K][N] in global.
// lane L holds col n = tile*16 + (L&15); half h -> k = kc*32 + (L>>4)*16 + h
__device__ __forceinline__ v16h load_b_frag_global(const float* __restrict__ W,
                                                   int K, int N, int lane,
                                                   int kc, int tile) {
    const int n = tile * 16 + (lane & 15);
    const int kbase = kc * 32 + ((lane >> 4) * 16);
    v16h b;
#pragma unroll
    for (int h = 0; h < 16; ++h) {
        int k = kbase + h;
        float v = (k < K && n < N) ? W[(size_t)k * N + n] : 0.0f;
        b[h] = (_Float16)v;
    }
    return b;
}

// C/D f32 accumulator store: VGPR r, lane L -> m = r + 8*(L>>4), n = L&15
__device__ __forceinline__ void store_acc(float* Zs, int strideZ, int lane,
                                          int tile, v8f acc) {
    const int n = tile * 16 + (lane & 15);
    const int mshift = (lane >> 4) * 8;
#pragma unroll
    for (int r = 0; r < 8; ++r) {
        Zs[(mshift + r) * strideZ + n] = acc[r];
    }
}

#define WMMA_F16(a, b, c) \
    __builtin_amdgcn_wmma_f32_16x16x32_f16(false, (a), false, (b), (short)0, (c), false, false)

// ---------------------------------------------------------------------------
// Kernel 1: fused 2-layer LSTM scan (one direction per blockIdx.y).
// Each workgroup: 16 batch rows, 8 wave32s, state resident in LDS for all
// 300 steps. Weights live in VGPRs as pre-built WMMA B fragments. The x tile
// for step s+1 is prefetched into registers while step s computes.
// ---------------------------------------------------------------------------
__global__ __launch_bounds__(256) void lstm_scan_kernel(
    const float* __restrict__ x,
    const float* __restrict__ kfw1, const float* __restrict__ bfw1,
    const float* __restrict__ kfw2, const float* __restrict__ bfw2,
    const float* __restrict__ kbw1, const float* __restrict__ bbw1,
    const float* __restrict__ kbw2, const float* __restrict__ bbw2,
    float* __restrict__ taps)   // [2][7][BATCH][H2C]
{
    const int dir  = blockIdx.y;
    const int b0   = blockIdx.x * 16;
    const int tid  = threadIdx.x;
    const int lane = tid & 31;
    const int wave = tid >> 5;

    const float* k1 = dir ? kbw1 : kfw1;
    const float* b1 = dir ? bbw1 : bfw1;
    const float* k2 = dir ? kbw2 : kfw2;
    const float* b2 = dir ? bbw2 : bfw2;

    __shared__ _Float16 A1[16][K1P];   // [x_t | h1 | pad]
    __shared__ _Float16 A2[16][K2P];   // [h1  | h2 | pad]
    __shared__ float    Z[16][N1T * 16];
    __shared__ float    C1s[16][H1C];
    __shared__ float    C2s[16][H2C];
    __shared__ float    bias1[N1];
    __shared__ float    bias2[N2];

    // ---- persistent weight fragments in VGPRs ----
    // layer1: wave w -> N-tile w; wave 0 additionally tile 8.
    // layer2: wave w -> N-tile w.
    const bool extra = (wave == 0);
    v16h W1a[3], W1b[3], W2f[3];
    v16h vzero = {};
#pragma unroll
    for (int kc = 0; kc < 3; ++kc) {
        W1a[kc] = load_b_frag_global(k1, IN1, N1, lane, kc, wave);
        W1b[kc] = extra ? load_b_frag_global(k1, IN1, N1, lane, kc, 8) : vzero;
        W2f[kc] = load_b_frag_global(k2, IN2, N2, lane, kc, wave);
    }

    // ---- init state / pads / biases ----
    for (int i = tid; i < 16 * K1P; i += 256) {
        A1[i / K1P][i % K1P] = (_Float16)0.0f;
        A2[i / K1P][i % K1P] = (_Float16)0.0f;
    }
    for (int i = tid; i < 16 * H1C; i += 256) C1s[i / H1C][i % H1C] = 0.0f;
    for (int i = tid; i < 16 * H2C; i += 256) C2s[i / H2C][i % H2C] = 0.0f;
    for (int i = tid; i < N1; i += 256) bias1[i] = b1[i];
    for (int i = tid; i < N2; i += 256) bias2[i] = b2[i];

    // x prefetch helpers (3 elements per thread cover 16*39 = 624)
    auto fetch_x = [&](int tt, float* dst) {
#pragma unroll
        for (int u = 0; u < XPF; ++u) {
            int i = tid + u * 256;
            if (i < 16 * FF) {
                int r = i / FF, f = i % FF;
                dst[u] = x[(size_t)(b0 + r) * (TT * FF) + tt * FF + f];
            }
        }
    };
    auto commit_x = [&](const float* src) {
#pragma unroll
        for (int u = 0; u < XPF; ++u) {
            int i = tid + u * 256;
            if (i < 16 * FF) A1[i / FF][i % FF] = (_Float16)src[u];
        }
    };

    // stage x for step 0
    float xcur[XPF];
    fetch_x(dir ? (TT - 1) : 0, xcur);
    commit_x(xcur);
    __syncthreads();

    for (int s = 0; s < TT; ++s) {
        const int tt = dir ? (TT - 1 - s) : s;   // time index into x

        // issue next step's x loads now; they complete behind the WMMA work
        float xnext[XPF];
        if (s + 1 < TT) fetch_x(dir ? (TT - 2 - s) : (s + 1), xnext);

        // ---- layer 1: z1 = [x_t, h1] @ k1 ----
        {
            v16h a0 = load_a_frag(&A1[0][0], lane, 0, K1P);
            v16h a1 = load_a_frag(&A1[0][0], lane, 1, K1P);
            v16h a2 = load_a_frag(&A1[0][0], lane, 2, K1P);
            v8f acc = {};
            acc = WMMA_F16(a0, W1a[0], acc);
            acc = WMMA_F16(a1, W1a[1], acc);
            acc = WMMA_F16(a2, W1a[2], acc);
            store_acc(&Z[0][0], N1T * 16, lane, wave, acc);
            if (extra) {  // wave-uniform branch: EXEC stays all-1
                v8f acc2 = {};
                acc2 = WMMA_F16(a0, W1b[0], acc2);
                acc2 = WMMA_F16(a1, W1b[1], acc2);
                acc2 = WMMA_F16(a2, W1b[2], acc2);
                store_acc(&Z[0][0], N1T * 16, lane, 8, acc2);
            }
        }
        __syncthreads();

        // ---- layer 1 gates (i, j, f, o); also commit prefetched x_{s+1} ----
        for (int i = tid; i < 16 * H1C; i += 256) {
            int r = i / H1C, n = i % H1C;
            float zi = Z[r][n]           + bias1[n];
            float zj = Z[r][H1C + n]     + bias1[H1C + n];
            float zf = Z[r][2 * H1C + n] + bias1[2 * H1C + n];
            float zo = Z[r][3 * H1C + n] + bias1[3 * H1C + n];
            float c = sigf(zf + 1.0f) * C1s[r][n] + sigf(zi) * tanhfast(zj);
            float h = sigf(zo) * tanhfast(c);
            C1s[r][n] = c;
            _Float16 hh = (_Float16)h;
            A1[r][FF + n] = hh;   // recurrent input for next step
            A2[r][n]      = hh;   // layer-2 input this step
        }
        if (s + 1 < TT) commit_x(xnext);   // A1 x-region free until next step
        __syncthreads();

        // ---- layer 2: z2 = [h1_t, h2] @ k2 ----
        {
            v16h a0 = load_a_frag(&A2[0][0], lane, 0, K2P);
            v16h a1 = load_a_frag(&A2[0][0], lane, 1, K2P);
            v16h a2 = load_a_frag(&A2[0][0], lane, 2, K2P);
            v8f acc = {};
            acc = WMMA_F16(a0, W2f[0], acc);
            acc = WMMA_F16(a1, W2f[1], acc);
            acc = WMMA_F16(a2, W2f[2], acc);
            store_acc(&Z[0][0], N1T * 16, lane, wave, acc);
        }
        __syncthreads();

        // ---- layer 2 gates + tap write ----
        const int tap = tap_index(tt);
        for (int i = tid; i < 16 * H2C; i += 256) {
            int r = i / H2C, n = i % H2C;
            float zi = Z[r][n]           + bias2[n];
            float zj = Z[r][H2C + n]     + bias2[H2C + n];
            float zf = Z[r][2 * H2C + n] + bias2[2 * H2C + n];
            float zo = Z[r][3 * H2C + n] + bias2[3 * H2C + n];
            float c = sigf(zf + 1.0f) * C2s[r][n] + sigf(zi) * tanhfast(zj);
            float h = sigf(zo) * tanhfast(c);
            C2s[r][n] = c;
            A2[r][H1C + n] = (_Float16)h;
            if (tap >= 0) {
                taps[(((size_t)dir * NTAPS + tap) * BATCH + (b0 + r)) * H2C + n] = h;
            }
        }
        __syncthreads();
    }
}

// ---------------------------------------------------------------------------
// Kernel 2: 3-head MLP on the tapped features.
// fc1 (420 -> 240) via WMMA; fc2/fc3 small scalar.
// Grid: (BATCH/16, 3 heads), 256 threads.
// ---------------------------------------------------------------------------
__global__ __launch_bounds__(256) void mlp_head_kernel(
    const float* __restrict__ taps,   // [2][7][BATCH][H2C]
    const float* __restrict__ hw1, const float* __restrict__ hb1,
    const float* __restrict__ hw2, const float* __restrict__ hb2,
    const float* __restrict__ hw3, const float* __restrict__ hb3,
    float* __restrict__ out)          // [BATCH][3][2]
{
    const int head = blockIdx.y;
    const int b0   = blockIdx.x * 16;
    const int tid  = threadIdx.x;
    const int lane = tid & 31;
    const int wave = tid >> 5;

    __shared__ _Float16 Ain[16][FCINP];
    __shared__ float    H1s[16][FC1C];
    __shared__ float    H2s[16][FC2C];

    // gather fc_in: [tap0 fw(30) bw(30) | tap1 ... ], pad 420..447 with 0
    for (int i = tid; i < 16 * FCINP; i += 256) {
        int r = i / FCINP, c = i % FCINP;
        float v = 0.0f;
        if (c < FCIN) {
            int k = c / (2 * H2C);
            int cc = c % (2 * H2C);
            int d = (cc >= H2C) ? 1 : 0;
            int n = cc - d * H2C;
            v = taps[(((size_t)d * NTAPS + k) * BATCH + (b0 + r)) * H2C + n];
        }
        Ain[r][c] = (_Float16)v;
    }
    __syncthreads();

    // ---- fc1: [16 x 448] @ [420 x 240], 15 N-tiles over 8 waves ----
    const float* W1 = hw1 + (size_t)head * FCIN * FC1C;
    const float* B1 = hb1 + head * FC1C;
    {
        const int t0 = wave;
        const int t1 = wave + 8;
        const bool has2 = (wave < 7);   // 15 tiles total; wave-uniform
        v8f acc0 = {}, acc1 = {};
        for (int kc = 0; kc < FCINP / 32; ++kc) {
            v16h a = load_a_frag(&Ain[0][0], lane, kc, FCINP);
            v16h bf0 = load_b_frag_global(W1, FCIN, FC1C, lane, kc, t0);
            acc0 = WMMA_F16(a, bf0, acc0);
            if (has2) {
                v16h bf1 = load_b_frag_global(W1, FCIN, FC1C, lane, kc, t1);
                acc1 = WMMA_F16(a, bf1, acc1);
            }
        }
        const int n0 = t0 * 16 + (lane & 15);
        const int mshift = (lane >> 4) * 8;
#pragma unroll
        for (int r = 0; r < 8; ++r) {
            float v = acc0[r] + B1[n0];
            H1s[mshift + r][n0] = v > 0.0f ? v : 0.0f;
        }
        if (has2) {
            const int n1 = t1 * 16 + (lane & 15);
#pragma unroll
            for (int r = 0; r < 8; ++r) {
                float v = acc1[r] + B1[n1];
                H1s[mshift + r][n1] = v > 0.0f ? v : 0.0f;
            }
        }
    }
    __syncthreads();

    // ---- fc2: 240 -> 80, relu ----
    const float* W2 = hw2 + (size_t)head * FC1C * FC2C;
    const float* B2 = hb2 + head * FC2C;
    for (int i = tid; i < 16 * FC2C; i += 256) {
        int r = i / FC2C, o = i % FC2C;
        float acc = B2[o];
        for (int kk = 0; kk < FC1C; ++kk)
            acc += H1s[r][kk] * W2[(size_t)kk * FC2C + o];
        H2s[r][o] = acc > 0.0f ? acc : 0.0f;
    }
    __syncthreads();

    // ---- fc3: 80 -> 2, output [b][head][o] ----
    const float* W3 = hw3 + (size_t)head * FC2C * OUTC;
    const float* B3 = hb3 + head * OUTC;
    for (int i = tid; i < 16 * OUTC; i += 256) {
        int r = i / OUTC, o = i % OUTC;
        float acc = B3[o];
        for (int kk = 0; kk < FC2C; ++kk)
            acc += H2s[r][kk] * W3[kk * OUTC + o];
        out[(size_t)(b0 + r) * (3 * OUTC) + head * OUTC + o] = acc;
    }
}

// ---------------------------------------------------------------------------
// Launch
// ---------------------------------------------------------------------------
extern "C" void kernel_launch(void* const* d_in, const int* in_sizes, int n_in,
                              void* d_out, int out_size, void* d_ws, size_t ws_size,
                              hipStream_t stream) {
    const float* x    = (const float*)d_in[0];
    const float* kfw1 = (const float*)d_in[1];
    const float* bfw1 = (const float*)d_in[2];
    const float* kfw2 = (const float*)d_in[3];
    const float* bfw2 = (const float*)d_in[4];
    const float* kbw1 = (const float*)d_in[5];
    const float* bbw1 = (const float*)d_in[6];
    const float* kbw2 = (const float*)d_in[7];
    const float* bbw2 = (const float*)d_in[8];
    const float* hw1  = (const float*)d_in[9];
    const float* hb1  = (const float*)d_in[10];
    const float* hw2  = (const float*)d_in[11];
    const float* hb2  = (const float*)d_in[12];
    const float* hw3  = (const float*)d_in[13];
    const float* hb3  = (const float*)d_in[14];

    float* taps = (float*)d_ws;          // 2*7*4096*30 floats = 6.9 MB
    float* out  = (float*)d_out;

    lstm_scan_kernel<<<dim3(BATCH / 16, 2), 256, 0, stream>>>(
        x, kfw1, bfw1, kfw2, bfw2, kbw1, bbw1, kbw2, bbw2, taps);

    mlp_head_kernel<<<dim3(BATCH / 16, 3), 256, 0, stream>>>(
        taps, hw1, hb1, hw2, hb2, hw3, hb3, out);
}
